// TcMoeRouter_59047210385431
// MI455X (gfx1250) — matrix operand order repaired and checked by reading, not compiled, not run on hardware
//
#include <hip/hip_runtime.h>
#include <math.h>

#define N_TOKENS  16384
#define HIDDEN    4096
#define N_EXPERTS 128
#define TOP_K     8

typedef __attribute__((ext_vector_type(16))) __bf16 v16bf;
typedef __attribute__((ext_vector_type(8)))  float  v8f;
typedef __attribute__((ext_vector_type(4)))  float  f32x4;

// One block = 16 tokens x 128 experts. 8 wave32s; wave w owns experts [16w,16w+16).
// GEMM: v_wmma_f32_16x16x32_bf16, K stepped by 32, fp32->bf16 converted in-flight.
__global__ __launch_bounds__(256)
void moe_router_kernel(const float* __restrict__ x,
                       const float* __restrict__ rw,
                       const float* __restrict__ thr,
                       float* __restrict__ gates_out,
                       float* __restrict__ idx_out,
                       float* __restrict__ w_out) {
    __shared__ float sLogits[16 * N_EXPERTS];
    __shared__ float sGates [16 * N_EXPERTS];
    __shared__ float sThr[N_EXPERTS];

    const int tid  = threadIdx.x;
    const int lane = tid & 31;
    const int wave = tid >> 5;
    const int tok0 = blockIdx.x * 16;

    if (tid < N_EXPERTS) sThr[tid] = fabsf(thr[tid]);

    // ---------------- WMMA GEMM ----------------
    const int  eBase = wave * 16;
    const int  mIdx  = lane & 15;
    const bool hi    = (lane & 16) != 0;

    const float* xrow = x  + (size_t)(tok0  + mIdx) * HIDDEN;
    const float* wrow = rw + (size_t)(eBase + mIdx) * HIDDEN;

    v8f c = {};
    for (int k0 = 0; k0 < HIDDEN; k0 += 32) {
        // A fragment (16x32 bf16): lane<16 -> K in [0,8)+[16,24); lane>=16 -> [8,16)+[24,32)
        const int ka = k0 + (hi ? 8 : 0);
        f32x4 a0 = *(const f32x4*)(xrow + ka);
        f32x4 a1 = *(const f32x4*)(xrow + ka + 4);
        f32x4 a2 = *(const f32x4*)(xrow + ka + 16);
        f32x4 a3 = *(const f32x4*)(xrow + ka + 20);
        // B fragment (32x16 bf16): lane<16 -> K [0,8)+[16,24); lane>=16 -> [8,16)+[24,32); N = lane&15
        const int kb = k0 + (hi ? 8 : 0);
        f32x4 b0 = *(const f32x4*)(wrow + kb);
        f32x4 b1 = *(const f32x4*)(wrow + kb + 4);
        f32x4 b2 = *(const f32x4*)(wrow + kb + 16);
        f32x4 b3 = *(const f32x4*)(wrow + kb + 20);

        v16bf a, b;
        #pragma unroll
        for (int i = 0; i < 4; ++i) {
            a[i]      = (__bf16)a0[i];
            a[4 + i]  = (__bf16)a1[i];
            a[8 + i]  = (__bf16)a2[i];
            a[12 + i] = (__bf16)a3[i];
            b[i]      = (__bf16)b0[i];
            b[4 + i]  = (__bf16)b1[i];
            b[8 + i]  = (__bf16)b2[i];
            b[12 + i] = (__bf16)b3[i];
        }
        c = __builtin_amdgcn_wmma_f32_16x16x32_bf16(false, a, false, b,
                                                    (short)0, c, false, false);
    }

    // C layout: N = lane&15, M = vgpr + (lane>=16 ? 8 : 0)
    {
        const int n    = eBase + mIdx;
        const int mOff = hi ? 8 : 0;
        #pragma unroll
        for (int r = 0; r < 8; ++r)
            sLogits[(mOff + r) * N_EXPERTS + n] = c[r];
    }
    __syncthreads();

    // ---------------- Gating: 1 thread per token ----------------
    if (tid < 16) {
        const float* L = &sLogits[tid * N_EXPERTS];
        float*       G = &sGates [tid * N_EXPERTS];

        float mx = -INFINITY;
        for (int e = 0; e < N_EXPERTS; ++e) mx = fmaxf(mx, L[e]);
        float denom = 0.f;
        for (int e = 0; e < N_EXPERTS; ++e) denom += __expf(L[e] - mx);
        const float inv_denom = 1.f / denom;

        int nz = 0;
        for (int e = 0; e < N_EXPERTS; ++e) {
            const float l = L[e], t = sThr[e];
            const float g = (l > t) ? 1.f : ((l < -t) ? -1.f : 0.f);
            G[e] = g;
            nz += (g != 0.f);
        }
        if (nz == 0) {
            // fallback: top-8 by logit == top-8 by softmax prob; those gates -> +1
            float tv[TOP_K]; int ti[TOP_K];
            #pragma unroll
            for (int j = 0; j < TOP_K; ++j) { tv[j] = -INFINITY; ti[j] = 0; }
            for (int e = 0; e < N_EXPERTS; ++e) {
                const float v = L[e];
                if (v > tv[TOP_K - 1]) {
                    int j = TOP_K - 1;
                    while (j > 0 && v > tv[j - 1]) { tv[j] = tv[j-1]; ti[j] = ti[j-1]; --j; }
                    tv[j] = v; ti[j] = e;
                }
            }
            for (int j = 0; j < TOP_K; ++j) G[ti[j]] = 1.f;
        }

        // top-8 by |gate * prob|, keep sign, normalize
        float av[TOP_K], sv[TOP_K]; int si[TOP_K];
        #pragma unroll
        for (int j = 0; j < TOP_K; ++j) { av[j] = -1.f; sv[j] = 0.f; si[j] = 0; }
        for (int e = 0; e < N_EXPERTS; ++e) {
            const float w = G[e] * __expf(L[e] - mx) * inv_denom;
            const float a = fabsf(w);
            if (a > av[TOP_K - 1]) {
                int j = TOP_K - 1;
                while (j > 0 && a > av[j - 1]) {
                    av[j] = av[j-1]; sv[j] = sv[j-1]; si[j] = si[j-1]; --j;
                }
                av[j] = a; sv[j] = w; si[j] = e;
            }
        }
        float s = 0.f;
        for (int j = 0; j < TOP_K; ++j) s += fabsf(sv[j]);
        const float invs = 1.f / fmaxf(s, 1e-9f);

        const int tok = tok0 + tid;
        for (int j = 0; j < TOP_K; ++j) {
            idx_out[(size_t)tok * TOP_K + j] = (float)si[j];
            w_out  [(size_t)tok * TOP_K + j] = sv[j] * invs;
        }
    }
    __syncthreads();

    // coalesced gates writeback (2048 floats by 256 threads)
    for (int i = tid; i < 16 * N_EXPERTS; i += 256) {
        const int m = i >> 7, e = i & 127;
        gates_out[(size_t)(tok0 + m) * N_EXPERTS + e] = sGates[i];
    }
}

extern "C" void kernel_launch(void* const* d_in, const int* in_sizes, int n_in,
                              void* d_out, int out_size, void* d_ws, size_t ws_size,
                              hipStream_t stream) {
    const float* x   = (const float*)d_in[0];
    const float* rw  = (const float*)d_in[1];
    const float* thr = (const float*)d_in[2];

    float* out       = (float*)d_out;
    float* gates_out = out;
    float* idx_out   = gates_out + (size_t)N_TOKENS * N_EXPERTS;
    float* w_out     = idx_out   + (size_t)N_TOKENS * TOP_K;

    dim3 grid(N_TOKENS / 16);
    dim3 block(256);
    hipLaunchKernelGGL(moe_router_kernel, grid, block, 0, stream,
                       x, rw, thr, gates_out, idx_out, w_out);
}
